// MultiHeadAttentionLayer_71244917506501
// MI455X (gfx1250) — matrix operand order repaired and checked by reading, not compiled, not run on hardware
//
#include <hip/hip_runtime.h>
#include <hip/hip_bf16.h>

typedef __attribute__((ext_vector_type(16))) __bf16 v16bf;
typedef __attribute__((ext_vector_type(8)))  __bf16 v8bf;
typedef __attribute__((ext_vector_type(8)))  float  v8f;
typedef __attribute__((ext_vector_type(4)))  float  f32x4;

#define HID 512
#define NHEAD 8
#define HDIM 64
#define BATCH 4
#define SEQ 2048
#define MTOT (BATCH * SEQ)   // 8192 rows for projections

// ---------------------------------------------------------------------------
// DPP16 cross-lane reductions over 16-lane rows (wave32 = 2 independent rows,
// matching the C-layout halves).  No LDS pipe, pure VALU.
// ---------------------------------------------------------------------------
template <int CTRL>
__device__ __forceinline__ float dpp_swap(float x) {
  return __int_as_float(
      __builtin_amdgcn_mov_dpp(__float_as_int(x), CTRL, 0xf, 0xf, true));
}
__device__ __forceinline__ float redmax16(float x) {
  x = fmaxf(x, dpp_swap<0xB1>(x));   // quad_perm [1,0,3,2]  (xor 1)
  x = fmaxf(x, dpp_swap<0x4E>(x));   // quad_perm [2,3,0,1]  (xor 2)
  x = fmaxf(x, dpp_swap<0x141>(x));  // row_half_mirror      (xor 4 eff.)
  x = fmaxf(x, dpp_swap<0x140>(x));  // row_mirror           (xor 8 eff.)
  return x;
}
__device__ __forceinline__ float redsum16(float x) {
  x += dpp_swap<0xB1>(x);
  x += dpp_swap<0x4E>(x);
  x += dpp_swap<0x141>(x);
  x += dpp_swap<0x140>(x);
  return x;
}

// ---------------------------------------------------------------------------
// A-operand (16x32 bf16, M x K): lane l<16 holds row m=l, halves 0-7 = K 0..7,
// halves 8-15 = K 16..23; lanes 16-31 hold K 8..15 / 24..31. Source row-major.
// ---------------------------------------------------------------------------
__device__ __forceinline__ v16bf load_a_tile(const __bf16* base, int ld, int r0,
                                             int c0, int lane) {
  const __bf16* p = base + (size_t)(r0 + (lane & 15)) * ld + c0 + ((lane & 16) ? 8 : 0);
  v8bf lo = *(const v8bf*)p;        // K off0 .. off0+7
  v8bf hi = *(const v8bf*)(p + 16); // K off0+16 .. off0+23
  v16bf a;
#pragma unroll
  for (int i = 0; i < 8; ++i) { a[i] = lo[i]; a[i + 8] = hi[i]; }
  return a;
}

__device__ __forceinline__ v16bf load_a_tile_f32(const float* base, int ld, int r0,
                                                 int c0, int lane) {
  const float* p = base + (size_t)(r0 + (lane & 15)) * ld + c0 + ((lane & 16) ? 8 : 0);
  f32x4 a0 = *(const f32x4*)p;
  f32x4 a1 = *(const f32x4*)(p + 4);
  f32x4 b0 = *(const f32x4*)(p + 16);
  f32x4 b1 = *(const f32x4*)(p + 20);
  v16bf a;
#pragma unroll
  for (int i = 0; i < 4; ++i) {
    a[i]      = (__bf16)a0[i];
    a[i + 4]  = (__bf16)a1[i];
    a[i + 8]  = (__bf16)b0[i];
    a[i + 12] = (__bf16)b1[i];
  }
  return a;
}

// ---------------------------------------------------------------------------
// B-operand (32x16 bf16, K x N), source stored as B^T row-major [N][K]:
// lane l holds column n = l&15; halves 0-7 = K off0..off0+7, halves 8-15 =
// K off0+16..off0+23, off0 = c0 + (lane<16 ? 0 : 8).
// ---------------------------------------------------------------------------
__device__ __forceinline__ v16bf load_b_tile(const __bf16* baseT, int ld, int n0,
                                             int c0, int lane) {
  const __bf16* p = baseT + (size_t)(n0 + (lane & 15)) * ld + c0 + ((lane & 16) ? 8 : 0);
  v8bf lo = *(const v8bf*)p;
  v8bf hi = *(const v8bf*)(p + 16);
  v16bf b;
#pragma unroll
  for (int i = 0; i < 8; ++i) { b[i] = lo[i]; b[i + 8] = hi[i]; }
  return b;
}

__device__ __forceinline__ v8f wmma_bf16(v16bf a, v16bf b, v8f c) {
  return __builtin_amdgcn_wmma_f32_16x16x32_bf16(false, a, false, b, (short)0, c,
                                                 false, false);
}

// ---------------------------------------------------------------------------
// Kernel 1: W[in][out] f32 -> WT[out][in] bf16
// ---------------------------------------------------------------------------
__global__ void transpose_w_kernel(const float* __restrict__ W, __bf16* __restrict__ WT) {
  int idx = blockIdx.x * blockDim.x + threadIdx.x;  // 0 .. 512*512-1
  int n = idx >> 9;
  int k = idx & 511;
  WT[idx] = (__bf16)W[k * HID + n];
}

// ---------------------------------------------------------------------------
// Accumulator store (16x16 f32 C-layout).
// MODE 0: bf16 [B,H,S,64]   MODE 1: bf16 [B,H,64,S]   MODE 2: f32 [r][n]
// ---------------------------------------------------------------------------
template <int MODE>
__device__ __forceinline__ void store_acc(v8f c, int row0, int n0,
                                          const float* __restrict__ bias,
                                          void* __restrict__ dst, int lane) {
  int n = n0 + (lane & 15);
  float bn = bias[n];
  int mbase = (lane & 16) ? 8 : 0;
  if (MODE == 2) {
    float* out = (float*)dst;
#pragma unroll
    for (int i = 0; i < 8; ++i)
      out[(size_t)(row0 + mbase + i) * HID + n] = c[i] + bn;
  } else {
    __bf16* out = (__bf16*)dst;
    int h = n >> 6, d = n & 63;
#pragma unroll
    for (int i = 0; i < 8; ++i) {
      int r = row0 + mbase + i;
      int bb = r >> 11;          // / SEQ
      int ss = r & (SEQ - 1);
      float val = c[i] + bn;
      size_t off = (MODE == 0)
                       ? ((((size_t)bb * NHEAD + h) * SEQ + ss) * HDIM + d)
                       : ((((size_t)bb * NHEAD + h) * HDIM + d) * SEQ + ss);
      out[off] = (__bf16)val;
    }
  }
}

// ---------------------------------------------------------------------------
// GEMM  out[r][n] = sum_k X[r][k] * WT[n][k] + bias[n]
// One wave computes a 32x32 output tile (4 accumulators, 2 A + 2 B loads per
// K-step -> 16 FLOP/B from L2).  4096 waves total.
// ---------------------------------------------------------------------------
template <int XBF16, int MODE>
__global__ __launch_bounds__(128) void proj_gemm_kernel(
    const void* __restrict__ X, const __bf16* __restrict__ WT,
    const float* __restrict__ bias, void* __restrict__ dst) {
  int lane = threadIdx.x & 31;
  int wid = threadIdx.x >> 5;
  int tile = blockIdx.x * 4 + wid;  // 4096 tiles = 256 (M/32) x 16 (N/32)
  int mt = tile >> 4;
  int nt = tile & 15;
  int row0 = mt * 32;
  int n0 = nt * 32;

  v8f c00 = {0.f,0.f,0.f,0.f,0.f,0.f,0.f,0.f};
  v8f c01 = c00, c10 = c00, c11 = c00;

  for (int k = 0; k < HID; k += 32) {
    v16bf a0, a1;
    if (XBF16) {
      a0 = load_a_tile((const __bf16*)X, HID, row0, k, lane);
      a1 = load_a_tile((const __bf16*)X, HID, row0 + 16, k, lane);
    } else {
      a0 = load_a_tile_f32((const float*)X, HID, row0, k, lane);
      a1 = load_a_tile_f32((const float*)X, HID, row0 + 16, k, lane);
    }
    v16bf b0 = load_b_tile(WT, HID, n0, k, lane);
    v16bf b1 = load_b_tile(WT, HID, n0 + 16, k, lane);
    c00 = wmma_bf16(a0, b0, c00);
    c01 = wmma_bf16(a0, b1, c01);
    c10 = wmma_bf16(a1, b0, c10);
    c11 = wmma_bf16(a1, b1, c11);
  }

  store_acc<MODE>(c00, row0,      n0,      bias, dst, lane);
  store_acc<MODE>(c01, row0,      n0 + 16, bias, dst, lane);
  store_acc<MODE>(c10, row0 + 16, n0,      bias, dst, lane);
  store_acc<MODE>(c11, row0 + 16, n0 + 16, bias, dst, lane);
}

// ---------------------------------------------------------------------------
// Flash attention, one wave per (b,h, 16-row query tile).
// Q,K bf16 [B,H,S,64]; VT bf16 [B,H,64,S]; ctx bf16 [B,S,512].
// ---------------------------------------------------------------------------
__global__ __launch_bounds__(128) void attn_kernel(
    const __bf16* __restrict__ qbf, const __bf16* __restrict__ kbf,
    const __bf16* __restrict__ vtbf, __bf16* __restrict__ ctx) {
  __shared__ __align__(16) __bf16 lds_p[4][16][32];

  int lane = threadIdx.x & 31;
  int wid = threadIdx.x >> 5;
  int gw = blockIdx.x * 4 + wid;  // 4096 waves = 32 bh * 128 qtiles
  int qt = gw & 127;
  int bh = gw >> 7;

  const __bf16* qbase = qbf + (size_t)bh * SEQ * HDIM;
  const __bf16* kbase = kbf + (size_t)bh * SEQ * HDIM;
  const __bf16* vbase = vtbf + (size_t)bh * HDIM * SEQ;

  v16bf aq0 = load_a_tile(qbase, HDIM, qt * 16, 0, lane);
  v16bf aq1 = load_a_tile(qbase, HDIM, qt * 16, 32, lane);

  const float scale = 0.044194173824159216f;  // 1/sqrt(512)
  float rmax[8], rsum[8];
  v8f o0 = {0.f,0.f,0.f,0.f,0.f,0.f,0.f,0.f};
  v8f o1 = o0, o2 = o0, o3 = o0;
#pragma unroll
  for (int i = 0; i < 8; ++i) { rmax[i] = -3.0e38f; rsum[i] = 0.f; }

  for (int kb = 0; kb < SEQ; kb += 32) {
    // prefetch next iteration's K rows and V columns (global_prefetch_b8)
    if (kb + 32 < SEQ) {
      __builtin_prefetch(kbase + (size_t)(kb + 32 + lane) * HDIM, 0, 3);
      __builtin_prefetch(vbase + (size_t)lane * SEQ + kb + 32, 0, 3);
      __builtin_prefetch(vbase + (size_t)(lane + 32) * SEQ + kb + 32, 0, 3);
    }

    // issue all K B-tile loads, then the 4 score WMMAs
    v16bf bk00 = load_b_tile(kbase, HDIM, kb,      0,  lane);
    v16bf bk01 = load_b_tile(kbase, HDIM, kb,      32, lane);
    v16bf bk10 = load_b_tile(kbase, HDIM, kb + 16, 0,  lane);
    v16bf bk11 = load_b_tile(kbase, HDIM, kb + 16, 32, lane);
    v8f z = {0.f,0.f,0.f,0.f,0.f,0.f,0.f,0.f};
    v8f s0 = wmma_bf16(aq0, bk00, z);
    s0     = wmma_bf16(aq1, bk01, s0);
    v8f s1 = wmma_bf16(aq0, bk10, z);
    s1     = wmma_bf16(aq1, bk11, s1);

    // hoist V B-tile loads: latency hides under the softmax VALU work
    v16bf bv0 = load_b_tile(vbase, SEQ, 0,  kb, lane);
    v16bf bv1 = load_b_tile(vbase, SEQ, 16, kb, lane);
    v16bf bv2 = load_b_tile(vbase, SEQ, 32, kb, lane);
    v16bf bv3 = load_b_tile(vbase, SEQ, 48, kb, lane);

    // online softmax per row (row m = i + 8*(lane>=16); cols across 16 lanes)
#pragma unroll
    for (int i = 0; i < 8; ++i) {
      float x0 = s0[i] * scale;
      float x1 = s1[i] * scale;
      float t = redmax16(fmaxf(x0, x1));
      float nm = fmaxf(rmax[i], t);
      float alpha = __expf(rmax[i] - nm);
      rmax[i] = nm;
      float p0 = __expf(x0 - nm);
      float p1 = __expf(x1 - nm);
      float ps = redsum16(p0 + p1);
      rsum[i] = rsum[i] * alpha + ps;
      o0[i] *= alpha; o1[i] *= alpha; o2[i] *= alpha; o3[i] *= alpha;
      // stage probs in C-layout -> LDS [m][k]
      int m = ((lane & 16) ? 8 : 0) + i;
      int ncol = lane & 15;
      lds_p[wid][m][ncol] = (__bf16)p0;
      lds_p[wid][m][16 + ncol] = (__bf16)p1;
    }
    __syncthreads();

    // reload probs as A-operand (16x32) from LDS
    int pr = lane & 15;
    int pc = (lane & 16) ? 8 : 0;
    v8bf plo = *(const v8bf*)&lds_p[wid][pr][pc];
    v8bf phi = *(const v8bf*)&lds_p[wid][pr][pc + 16];
    v16bf ap;
#pragma unroll
    for (int i = 0; i < 8; ++i) { ap[i] = plo[i]; ap[i + 8] = phi[i]; }

    // O += P * V  (B operand from pre-transposed V: VT[d][s])
    o0 = wmma_bf16(ap, bv0, o0);
    o1 = wmma_bf16(ap, bv1, o1);
    o2 = wmma_bf16(ap, bv2, o2);
    o3 = wmma_bf16(ap, bv3, o3);
    __syncthreads();
  }

  // epilogue: normalize, write ctx bf16 [B,S,512]
  int ncol = lane & 15;
  int mbase = (lane & 16) ? 8 : 0;
  int b = bh >> 3, h = bh & 7;
#pragma unroll
  for (int i = 0; i < 8; ++i) {
    int srow = qt * 16 + mbase + i;
    float inv = 1.0f / rsum[i];
    size_t base = ((size_t)b * SEQ + srow) * HID + h * HDIM;
    ctx[base + 0  + ncol] = (__bf16)(o0[i] * inv);
    ctx[base + 16 + ncol] = (__bf16)(o1[i] * inv);
    ctx[base + 32 + ncol] = (__bf16)(o2[i] * inv);
    ctx[base + 48 + ncol] = (__bf16)(o3[i] * inv);
  }
}

// ---------------------------------------------------------------------------
extern "C" void kernel_launch(void* const* d_in, const int* in_sizes, int n_in,
                              void* d_out, int out_size, void* d_ws, size_t ws_size,
                              hipStream_t stream) {
  const float* q  = (const float*)d_in[0];
  const float* k  = (const float*)d_in[1];
  const float* v  = (const float*)d_in[2];
  const float* Wq = (const float*)d_in[3];
  const float* bq = (const float*)d_in[4];
  const float* Wk = (const float*)d_in[5];
  const float* bk = (const float*)d_in[6];
  const float* Wv = (const float*)d_in[7];
  const float* bv = (const float*)d_in[8];
  const float* Wo = (const float*)d_in[9];
  const float* bo = (const float*)d_in[10];

  char* ws = (char*)d_ws;
  const size_t WSZ = (size_t)HID * HID * sizeof(__bf16);      // 512 KB
  const size_t ASZ = (size_t)MTOT * HID * sizeof(__bf16);     // 8 MB
  __bf16* WqT  = (__bf16*)(ws);
  __bf16* WkT  = (__bf16*)(ws + WSZ);
  __bf16* WvT  = (__bf16*)(ws + 2 * WSZ);
  __bf16* WoT  = (__bf16*)(ws + 3 * WSZ);
  __bf16* qbf  = (__bf16*)(ws + 4 * WSZ);
  __bf16* kbf  = (__bf16*)(ws + 4 * WSZ + ASZ);
  __bf16* vt   = (__bf16*)(ws + 4 * WSZ + 2 * ASZ);
  __bf16* ctxb = (__bf16*)(ws + 4 * WSZ + 3 * ASZ);

  // 1) weight transpose+convert
  transpose_w_kernel<<<(HID * HID) / 256, 256, 0, stream>>>(Wq, WqT);
  transpose_w_kernel<<<(HID * HID) / 256, 256, 0, stream>>>(Wk, WkT);
  transpose_w_kernel<<<(HID * HID) / 256, 256, 0, stream>>>(Wv, WvT);
  transpose_w_kernel<<<(HID * HID) / 256, 256, 0, stream>>>(Wo, WoT);

  // 2) QKV projections (4096 tiles of 32x32, 4 waves/block)
  proj_gemm_kernel<0, 0><<<1024, 128, 0, stream>>>(q, WqT, bq, qbf);
  proj_gemm_kernel<0, 0><<<1024, 128, 0, stream>>>(k, WkT, bk, kbf);
  proj_gemm_kernel<0, 1><<<1024, 128, 0, stream>>>(v, WvT, bv, vt);

  // 3) flash attention (4096 waves)
  attn_kernel<<<1024, 128, 0, stream>>>(qbf, kbf, vt, ctxb);

  // 4) output projection -> f32 d_out
  proj_gemm_kernel<1, 2><<<1024, 128, 0, stream>>>(ctxb, WoT, bo, d_out);
}
